// ProductQuantizer_89601607729713
// MI455X (gfx1250) — matrix operand order repaired and checked by reading, not compiled, not run on hardware
//
#include <hip/hip_runtime.h>
#include <hip/hip_bf16.h>
#include <cstdint>

// Problem constants (fixed by the reference)
constexpr int H_ = 8, K_ = 1024, D_ = 128, B_ = 8192;
constexpr float DECAY_ = 0.99f, EPS_ = 1e-5f;

typedef __bf16 bf16x16 __attribute__((ext_vector_type(16)));
typedef float f32x8 __attribute__((ext_vector_type(8)));

union BfVec { bf16x16 v; unsigned short u[16]; };

__device__ __forceinline__ unsigned short f32_to_bf16_rne(float f) {
    unsigned int u = __float_as_uint(f);
    unsigned int r = u + 0x7FFFu + ((u >> 16) & 1u);
    return (unsigned short)(r >> 16);
}
__device__ __forceinline__ float bf16_bits_to_f32(unsigned short h) {
    return __uint_as_float(((unsigned int)h) << 16);
}

// LDS byte address of a __shared__ object (AS3 pointers are 32-bit offsets)
__device__ __forceinline__ unsigned lds_off(void* p) {
    return (unsigned)(uintptr_t)(__attribute__((address_space(3))) void*)p;
}

// Async global->LDS copy of 32 bytes (2x b128), GVS addressing, ASYNCcnt-tracked.
__device__ __forceinline__ void async_cp32(unsigned lds, uint64_t sbase, unsigned voff) {
    asm volatile("global_load_async_to_lds_b128 %0, %1, %2 offset:0\n\t"
                 "global_load_async_to_lds_b128 %0, %1, %2 offset:16"
                 :: "v"(lds), "v"(voff), "s"(sbase) : "memory");
}

// ---------------------------------------------------------------------------
// Pack codebooks [H,K,D] f32 into bf16 hi/lo in the wave32 WMMA B-operand
// layout. B (32-deep K x 16-wide N, 16-bit) layout used here, contiguous k (A matches): lane&15 = N,
// (lane>>4)*16 + slot = K within the 32-deep k-step; 16 bf16 "slots" per lane
// stored contiguously so the distance kernel reads two ds_load_b128/operand.
// Packed index: ((h*64 + nt)*4 + ks)*512 + lane*16 + slot
// ---------------------------------------------------------------------------
__global__ void pack_codebook_kernel(const float* __restrict__ cb,
                                     unsigned short* __restrict__ hi,
                                     unsigned short* __restrict__ lo) {
    int tid  = blockIdx.x * blockDim.x + threadIdx.x;  // 0 .. H*K*D-1
    int slot = tid & 15;
    int lane = (tid >> 4) & 31;
    int ks   = (tid >> 9) & 3;
    int nt   = (tid >> 11) & 63;
    int h    = tid >> 17;
    int n    = lane & 15;
    int g    = lane >> 4;
    int kdim = ks * 32 + g * 16 + slot;
    int code = nt * 16 + n;
    float f = cb[((size_t)h * K_ + code) * D_ + kdim];
    unsigned short hb = f32_to_bf16_rne(f);
    hi[tid] = hb;
    lo[tid] = f32_to_bf16_rne(f - bf16_bits_to_f32(hb));
}

__global__ void c2_kernel(const float* __restrict__ cb, float* __restrict__ c2) {
    int i = blockIdx.x * blockDim.x + threadIdx.x;  // H*K entries
    if (i >= H_ * K_) return;
    const float* p = cb + (size_t)i * D_;
    float s = 0.f;
    #pragma unroll 8
    for (int d = 0; d < D_; ++d) { float v = p[d]; s += v * v; }
    c2[i] = s;
}

// ---------------------------------------------------------------------------
// Distance + argmin. Block = 4 waves; each wave owns a 16-row A tile held in
// registers (contiguous k order, matching packed B). Packed B tiles are streamed
// into double-buffered LDS with GLOBAL_LOAD_ASYNC_TO_LDS_B128 (ASYNCcnt),
// hiding the global latency behind the current tile's 12 WMMAs.
// argmin uses d2' = |c|^2 - 2 z.c (|z|^2 is constant per row).
// ---------------------------------------------------------------------------
__global__ void __launch_bounds__(128)
dist_argmin_kernel(const float* __restrict__ z,
                   const unsigned short* __restrict__ cb_hi,
                   const unsigned short* __restrict__ cb_lo,
                   const float* __restrict__ c2,
                   int* __restrict__ out_idx) {
    __shared__ unsigned short s_b[2][4096];  // per buffer: [0..2047]=hi, [2048..4095]=lo
    __shared__ float s_c2[1024];

    const int h    = blockIdx.y;
    const int wave = threadIdx.x >> 5;
    const int lane = threadIdx.x & 31;
    const int g    = lane >> 4;
    const int m    = lane & 15;
    const int tile_base = blockIdx.x * 64 + wave * 16;
    const int row  = tile_base + m;

    // --- async staging setup; kick off tile 0 before doing any VALU work ---
    const uint64_t gb_hi = (uint64_t)(uintptr_t)(cb_hi + (size_t)h * 64 * 2048);
    const uint64_t gb_lo = (uint64_t)(uintptr_t)(cb_lo + (size_t)h * 64 * 2048);
    const unsigned lane_byte = threadIdx.x * 32;       // this thread's 32B slice of a 4KB block
    const unsigned lds_hi[2] = { lds_off(&s_b[0][0])    + lane_byte,
                                 lds_off(&s_b[1][0])    + lane_byte };
    const unsigned lds_lo[2] = { lds_off(&s_b[0][2048]) + lane_byte,
                                 lds_off(&s_b[1][2048]) + lane_byte };
    async_cp32(lds_hi[0], gb_hi, lane_byte);
    async_cp32(lds_lo[0], gb_lo, lane_byte);

    // stage this head's |c|^2 into LDS (read every iteration)
    for (int t = threadIdx.x; t < 1024; t += 128) s_c2[t] = c2[h * K_ + t];

    // A tile: lane group g holds K = g*16+[0..15] (slots 0-15) within each
    // 32-deep kstep, matching the packed B layout.
    BfVec a_hi[4], a_lo[4];
    const float* zrow = z + (size_t)row * (H_ * D_) + h * D_;
    #pragma unroll
    for (int ks = 0; ks < 4; ++ks) {
        #pragma unroll
        for (int half = 0; half < 2; ++half) {
            int kbase = ks * 32 + g * 16 + half * 8;
            float4 f0 = *(const float4*)(zrow + kbase);
            float4 f1 = *(const float4*)(zrow + kbase + 4);
            float fv[8] = {f0.x, f0.y, f0.z, f0.w, f1.x, f1.y, f1.z, f1.w};
            #pragma unroll
            for (int j = 0; j < 8; ++j) {
                unsigned short hb = f32_to_bf16_rne(fv[j]);
                a_hi[ks].u[half * 8 + j] = hb;
                a_lo[ks].u[half * 8 + j] = f32_to_bf16_rne(fv[j] - bf16_bits_to_f32(hb));
            }
        }
    }

    float run_min[8];
    int   run_idx[8];
    #pragma unroll
    for (int r = 0; r < 8; ++r) { run_min[r] = 3.4e38f; run_idx[r] = 0; }

    for (int nt = 0; nt < 64; ++nt) {
        const int buf = nt & 1;
        if (nt < 63) {
            // issue next tile's copies into the other buffer, then retire the
            // current tile's 4 requests (async loads complete in order)
            unsigned vo = (unsigned)((nt + 1) * 4096) + lane_byte;
            async_cp32(lds_hi[buf ^ 1], gb_hi, vo);
            async_cp32(lds_lo[buf ^ 1], gb_lo, vo);
            asm volatile("s_wait_asynccnt 0x4" ::: "memory");
        } else {
            asm volatile("s_wait_asynccnt 0x0" ::: "memory");
        }
        __syncthreads();   // all waves' slices of the current tile are in LDS

        f32x8 acc = {};
        const unsigned short* sb = &s_b[buf][0];
        #pragma unroll
        for (int ks = 0; ks < 4; ++ks) {
            BfVec bh, bl;
            #pragma unroll
            for (int j = 0; j < 16; ++j) {
                bh.u[j] = sb[ks * 512 + lane * 16 + j];
                bl.u[j] = sb[2048 + ks * 512 + lane * 16 + j];
            }
            // split-bf16 f32 emulation: zh*ch + zh*cl + zl*ch
            acc = __builtin_amdgcn_wmma_f32_16x16x32_bf16(false, a_hi[ks].v, false, bh.v,
                                                          (short)0, acc, false, false);
            acc = __builtin_amdgcn_wmma_f32_16x16x32_bf16(false, a_hi[ks].v, false, bl.v,
                                                          (short)0, acc, false, false);
            acc = __builtin_amdgcn_wmma_f32_16x16x32_bf16(false, a_lo[ks].v, false, bh.v,
                                                          (short)0, acc, false, false);
        }

        int n = nt * 16 + (lane & 15);
        float c2v = s_c2[n];
        #pragma unroll
        for (int r = 0; r < 8; ++r) {
            float d2 = c2v - 2.0f * acc[r];
            if (d2 < run_min[r]) { run_min[r] = d2; run_idx[r] = n; }
        }
        __syncthreads();   // done reading this buffer; safe to overwrite next iter
    }

    // reduce across the 16 lanes of each half (the N dimension of the C tile);
    // tie-break to the lowest index (jnp.argmin semantics)
    #pragma unroll
    for (int r = 0; r < 8; ++r) {
        float v = run_min[r];
        int   i = run_idx[r];
        #pragma unroll
        for (int off = 8; off >= 1; off >>= 1) {
            float ov = __shfl_xor(v, off, 32);
            int   oi = __shfl_xor(i, off, 32);
            if (ov < v || (ov == v && oi < i)) { v = ov; i = oi; }
        }
        if ((lane & 15) == 0) {
            int M = r + g * 8;   // C layout: lanes 0-15 -> M=r, 16-31 -> M=r+8
            out_idx[h * B_ + tile_base + M] = i;
        }
    }
}

// ---------------------------------------------------------------------------
// Gather codewords (z_q forward value == zq), exact f32 commit loss, EMA
// histogram + segment-sum via global float atomics. One wave per (h,b) row.
// ---------------------------------------------------------------------------
__global__ void __launch_bounds__(128)
gather_scatter_kernel(const float* __restrict__ z,
                      const float* __restrict__ cb,
                      const int* __restrict__ idx,
                      float* __restrict__ zq_out,
                      float* __restrict__ idx_out,
                      float* __restrict__ sums,
                      float* __restrict__ counts,
                      float* __restrict__ loss_acc) {
    int rowid = blockIdx.x * 4 + (threadIdx.x >> 5);  // h*B + b
    int lane  = threadIdx.x & 31;
    int h = rowid / B_;
    int b = rowid - h * B_;
    int code = idx[rowid];
    const float* zp = z + (size_t)b * (H_ * D_) + h * D_;
    const float* cp = cb + ((size_t)h * K_ + code) * D_;
    float* sp = sums + ((size_t)h * K_ + code) * D_;
    float* qp = zq_out + (size_t)b * (H_ * D_) + h * D_;
    float lsum = 0.f;
    #pragma unroll
    for (int t = 0; t < 4; ++t) {
        int d = lane + t * 32;
        float zv = zp[d], cv = cp[d];
        qp[d] = cv;
        float df = zv - cv;
        lsum += df * df;
        atomicAdd(sp + d, zv);
    }
    #pragma unroll
    for (int off = 16; off >= 1; off >>= 1) lsum += __shfl_xor(lsum, off, 32);
    if (lane == 0) {
        atomicAdd(loss_acc, lsum);
        atomicAdd(counts + h * K_ + code, 1.0f);
        idx_out[(size_t)b * H_ + h] = (float)code;
    }
}

__global__ void finalize_kernel(const float* __restrict__ ema_count,
                                const float* __restrict__ ema_weight,
                                const float* __restrict__ counts,
                                const float* __restrict__ sums,
                                const float* __restrict__ loss_acc,
                                float* __restrict__ out_ncb,
                                float* __restrict__ out_nc,
                                float* __restrict__ out_nw,
                                float* __restrict__ out_loss) {
    int i = blockIdx.x * blockDim.x + threadIdx.x;  // H*K*D elements
    if (i == 0) out_loss[0] = loss_acc[0] / (float)((size_t)H_ * B_ * D_);
    if (i >= H_ * K_ * D_) return;
    int hk = i / D_;
    float nc = DECAY_ * ema_count[hk] + (1.0f - DECAY_) * counts[hk];
    float nw = DECAY_ * ema_weight[i] + (1.0f - DECAY_) * sums[i];
    out_nw[i]  = nw;
    out_ncb[i] = nw / (nc + EPS_);
    if ((i % D_) == 0) out_nc[hk] = nc;
}

extern "C" void kernel_launch(void* const* d_in, const int* in_sizes, int n_in,
                              void* d_out, int out_size, void* d_ws, size_t ws_size,
                              hipStream_t stream) {
    const float* z          = (const float*)d_in[0];  // [B, H*D]
    const float* cb         = (const float*)d_in[1];  // [H, K, D]
    const float* ema_count  = (const float*)d_in[2];  // [H, K]
    const float* ema_weight = (const float*)d_in[3];  // [H, K, D]

    // Workspace layout (~8.3 MB):
    //   [0,2MB)        cb_hi  ushort[H*K*D]
    //   [2MB,4MB)      cb_lo  ushort[H*K*D]
    //   [4MB,+32KB)    c2     float[H*K]
    //   [+,+256KB)     idx    int[H*B]
    //   [+,+32KB)      counts float[H*K]      --+
    //   [+,+4MB)       sums   float[H*K*D]      | contiguous zero region
    //   [+,+4B)        loss   float[1]        --+
    char* ws = (char*)d_ws;
    unsigned short* cb_hi = (unsigned short*)ws;
    unsigned short* cb_lo = (unsigned short*)(ws + ((size_t)2 << 20));
    float* c2     = (float*)(ws + ((size_t)4 << 20));
    int*   idx    = (int*)  (ws + ((size_t)4 << 20) + ((size_t)32 << 10));
    float* counts = (float*)(ws + ((size_t)4 << 20) + ((size_t)288 << 10));
    float* sums   = (float*)(ws + ((size_t)4 << 20) + ((size_t)320 << 10));
    float* loss   = (float*)(ws + ((size_t)8 << 20) + ((size_t)320 << 10));

    float* o_zq   = (float*)d_out;                    // [B, H*D]
    float* o_idx  = o_zq + (size_t)B_ * H_ * D_;      // [B, H] (as float)
    float* o_loss = o_idx + (size_t)B_ * H_;          // scalar
    float* o_ncb  = o_loss + 1;                       // [H, K, D]
    float* o_nc   = o_ncb + (size_t)H_ * K_ * D_;     // [H, K]
    float* o_nw   = o_nc + (size_t)H_ * K_;           // [H, K, D]

    // zero counts + sums + loss (contiguous) — memset node is graph-capturable
    hipMemsetAsync(counts, 0, ((size_t)32 << 10) + ((size_t)4 << 20) + 4, stream);

    pack_codebook_kernel<<<(H_ * K_ * D_) / 256, 256, 0, stream>>>(cb, cb_hi, cb_lo);
    c2_kernel<<<(H_ * K_) / 256, 256, 0, stream>>>(cb, c2);

    dim3 dgrid(B_ / 64, H_);
    dist_argmin_kernel<<<dgrid, 128, 0, stream>>>(z, cb_hi, cb_lo, c2, idx);

    gather_scatter_kernel<<<(H_ * B_) / 4, 128, 0, stream>>>(z, cb, idx, o_zq, o_idx,
                                                             sums, counts, loss);

    finalize_kernel<<<(H_ * K_ * D_) / 256, 256, 0, stream>>>(ema_count, ema_weight,
                                                              counts, sums, loss,
                                                              o_ncb, o_nc, o_nw, o_loss);
}
